// MoEFFNBlock_24395414241304
// MI455X (gfx1250) — matrix operand order repaired and checked by reading, not compiled, not run on hardware
//
#include <hip/hip_runtime.h>
#include <hip/hip_bf16.h>

// ---------------------------------------------------------------------------
// MoE FFN block for MI455X (gfx1250, wave32, WMMA).
//
// v2: split gate_up and down GEMMs with M=128 x N=64 tiles to cut L2 weight
// traffic 8x vs the fused M=16 design; A tiles staged in LDS via CDNA5 async
// copy (global_load_async_to_lds_b128 + s_wait_asynccnt); bf16 WMMA with f32
// accumulation; deterministic slot-gather combine.
// ---------------------------------------------------------------------------

#define kDim     1024
#define kHid     2816
#define kE       8
#define kTokens  8192
#define kSlots   (kTokens * 2)     // exactly 2 slots per token
#define kMTile   128               // token rows per GEMM block
#define kMTiles  (kTokens / kMTile)    // 64 worst-case M tiles per expert
#define kNTile   64                // output features per GEMM block
#define kKChunk  256               // K elements staged in LDS per phase

typedef __bf16 bf16_t;
typedef __attribute__((ext_vector_type(16))) __bf16 v16bf;
typedef __attribute__((ext_vector_type(8)))  __bf16 v8bf;
typedef __attribute__((ext_vector_type(4)))  __bf16 v4bf;
typedef __attribute__((ext_vector_type(8)))  float  v8f;
typedef __attribute__((ext_vector_type(4)))  float  v4f;

union BF16Frag { v16bf v; v8bf h[2]; };

static __device__ __forceinline__ v8f wmma_bf16(const BF16Frag& a,
                                                const BF16Frag& b, v8f c) {
    return __builtin_amdgcn_wmma_f32_16x16x32_bf16(
        false, a.v, false, b.v, (short)0, c, false, false);
}

// Async copy of 16 bytes global -> LDS (CDNA5 GLOBAL_LOAD_ASYNC_TO_LDS_B128,
// GV addressing mode). Completion via s_wait_asynccnt + workgroup barrier.
static __device__ __forceinline__ void async_copy_b128(unsigned lds_addr,
                                                       const void* gaddr) {
    asm volatile("global_load_async_to_lds_b128 %0, %1, off"
                 :: "v"(lds_addr), "v"(gaddr) : "memory");
}
static __device__ __forceinline__ void async_wait_all() {
    asm volatile("s_wait_asynccnt 0" ::: "memory");
}

// ---------------------------------------------------------------------------
// Kernel 0: zero the per-expert counters.
// ---------------------------------------------------------------------------
__global__ void k_zero_counts(int* counts) {
    if (threadIdx.x < kE) counts[threadIdx.x] = 0;
}

// ---------------------------------------------------------------------------
// Kernel 1: f32 -> bf16 weight conversion (vectorized 4-wide, grid-stride).
// ---------------------------------------------------------------------------
__global__ void k_cvt_f32_bf16(const float* __restrict__ src,
                               bf16_t* __restrict__ dst, long n4) {
    long i      = (long)blockIdx.x * blockDim.x + threadIdx.x;
    long stride = (long)gridDim.x * blockDim.x;
    for (; i < n4; i += stride) {
        v4f v = *(const v4f*)(src + i * 4);
        v4bf o;
        o.x = (bf16_t)v.x; o.y = (bf16_t)v.y;
        o.z = (bf16_t)v.z; o.w = (bf16_t)v.w;
        *(v4bf*)(dst + i * 4) = o;
    }
}

// ---------------------------------------------------------------------------
// Kernel 2: fused RMSNorm + router (softmax, top-2, renorm).
// ---------------------------------------------------------------------------
__global__ __launch_bounds__(256) void k_rmsnorm_router(
    const float* __restrict__ x, const float* __restrict__ norm_w,
    const float* __restrict__ router_w, bf16_t* __restrict__ normed,
    int* __restrict__ topk_e, float* __restrict__ topk_w,
    int* __restrict__ counts) {
    const int t   = blockIdx.x;
    const int tid = threadIdx.x;
    const float* xr = x + (size_t)t * kDim;

    float v[4], ss = 0.f;
#pragma unroll
    for (int j = 0; j < 4; ++j) { v[j] = xr[tid * 4 + j]; ss += v[j] * v[j]; }

    __shared__ float red[256];
    red[tid] = ss;
    __syncthreads();
    for (int s = 128; s > 0; s >>= 1) {
        if (tid < s) red[tid] += red[tid + s];
        __syncthreads();
    }
    const float rs = rsqrtf(red[0] * (1.0f / kDim) + 1e-6f);

    float nv[4];
#pragma unroll
    for (int j = 0; j < 4; ++j) {
        nv[j] = v[j] * rs * norm_w[tid * 4 + j];
        normed[(size_t)t * kDim + tid * 4 + j] = (bf16_t)nv[j];
    }

    __shared__ float slog[kE];
    if (tid < kE) slog[tid] = 0.f;
    __syncthreads();
#pragma unroll
    for (int e = 0; e < kE; ++e) {
        const float* rw = router_w + (size_t)e * kDim + tid * 4;
        float p = nv[0] * rw[0] + nv[1] * rw[1] + nv[2] * rw[2] + nv[3] * rw[3];
        atomicAdd(&slog[e], p);
    }
    __syncthreads();

    if (tid == 0) {
        float mx = slog[0];
#pragma unroll
        for (int e = 1; e < kE; ++e) mx = fmaxf(mx, slog[e]);
        float pe[kE];
#pragma unroll
        for (int e = 0; e < kE; ++e) pe[e] = __expf(slog[e] - mx);
        int e0 = 0;
#pragma unroll
        for (int e = 1; e < kE; ++e) if (pe[e] > pe[e0]) e0 = e;
        int e1 = (e0 == 0) ? 1 : 0;
#pragma unroll
        for (int e = 0; e < kE; ++e)
            if (e != e0 && pe[e] > pe[e1]) e1 = e;
        float w0 = pe[e0], w1 = pe[e1];
        float inv = 1.0f / (w0 + w1);
        topk_e[t * 2 + 0] = e0; topk_e[t * 2 + 1] = e1;
        topk_w[t * 2 + 0] = w0 * inv; topk_w[t * 2 + 1] = w1 * inv;
        atomicAdd(&counts[e0], 1);
        atomicAdd(&counts[e1], 1);
    }
}

// ---------------------------------------------------------------------------
// Kernel 3: serial prefix scan over 8 experts -> contiguous slot ranges.
// ---------------------------------------------------------------------------
__global__ void k_scan_offsets(const int* __restrict__ counts,
                               int* __restrict__ offsets,
                               int* __restrict__ cursors) {
    if (threadIdx.x == 0) {
        int acc = 0;
        for (int e = 0; e < kE; ++e) {
            offsets[e] = acc;
            acc += counts[e];
            cursors[e] = 0;
        }
    }
}

// ---------------------------------------------------------------------------
// Kernel 4: scatter token -> slot.
// ---------------------------------------------------------------------------
__global__ void k_scatter(const int* __restrict__ topk_e,
                          int* __restrict__ cursors,
                          const int* __restrict__ offsets,
                          int* __restrict__ tok_idx,
                          int* __restrict__ slot_of) {
    int t = blockIdx.x * blockDim.x + threadIdx.x;
    if (t >= kTokens) return;
#pragma unroll
    for (int k = 0; k < 2; ++k) {
        int e    = topk_e[t * 2 + k];
        int pos  = atomicAdd(&cursors[e], 1);
        int slot = offsets[e] + pos;
        tok_idx[slot]       = t;
        slot_of[t * 2 + k]  = slot;
    }
}

// ---------------------------------------------------------------------------
// Kernel 5a: gate_up GEMM + silu.   hidden[slot][h] = silu(g)*u, bf16.
// Block = (expert e, M tile of 128 slots, N tile of 64 hidden features).
// 8 waves arranged 4(M) x 2(N); each wave owns 32 rows x 32 cols of gate AND
// up (8 f32 accumulators). A tile staged in LDS in 256-wide K chunks via
// async copy; B fragments read straight from L2-resident bf16 weights
// (per-lane contiguous K matches row-major [feature][dim] layout).
// ---------------------------------------------------------------------------
__global__ __launch_bounds__(256) void k_gateup_silu(
    const bf16_t* __restrict__ normed, const bf16_t* __restrict__ gu_w,
    const int* __restrict__ tok_idx, const int* __restrict__ counts,
    const int* __restrict__ offsets, bf16_t* __restrict__ hidden) {
    const int e   = blockIdx.z;
    const int mt  = blockIdx.y;
    const int nt  = blockIdx.x;
    const int cnt = counts[e];
    const int m0  = mt * kMTile;
    if (m0 >= cnt) return;
    const int base = offsets[e];
    const int mcnt = min(kMTile, cnt - m0);

    __shared__ bf16_t sA[kMTile * kKChunk];   // 128 x 256 bf16 = 64 KB
    __shared__ int    sTok[kMTile];

    const int tid = threadIdx.x;
    if (tid < kMTile)
        sTok[tid] = tok_idx[base + m0 + ((tid < mcnt) ? tid : 0)];
    __syncthreads();

    const int lane  = tid & 31;
    const int wave  = tid >> 5;
    const int waveM = wave >> 1;          // 0..3
    const int waveN = wave & 1;           // 0..1
    const int mrow  = lane & 15;
    const int khalf = lane >> 4;
    const int wm0   = waveM * 32;
    const int hg0   = nt * kNTile + waveN * 32;

    const bf16_t* guE = gu_w + (size_t)e * (2 * kHid) * kDim;
    const bf16_t* bG[2];
    const bf16_t* bU[2];
#pragma unroll
    for (int ni = 0; ni < 2; ++ni) {
        bG[ni] = guE + (size_t)(hg0 + ni * 16 + mrow) * kDim + khalf * 16;
        bU[ni] = guE + (size_t)(hg0 + kHid + ni * 16 + mrow) * kDim + khalf * 16;
    }

    // Async A-gather: each thread owns 256 B of one half row.
    const int arow  = tid >> 1;
    const int ahalf = tid & 1;
    const bf16_t* aSrcRow =
        normed + (size_t)sTok[arow] * kDim + ahalf * (kKChunk / 2);
    const unsigned aLds =
        (unsigned)(uintptr_t)(&sA[arow * kKChunk + ahalf * (kKChunk / 2)]);

    v8f accG[2][2] = {{{}, {}}, {{}, {}}};
    v8f accU[2][2] = {{{}, {}}, {{}, {}}};

    for (int kc = 0; kc < kDim; kc += kKChunk) {
        const bf16_t* src = aSrcRow + kc;
#pragma unroll
        for (int i = 0; i < 16; ++i)
            async_copy_b128(aLds + i * 16, src + i * 8);
        async_wait_all();
        __syncthreads();

        for (int kk = 0; kk < kKChunk; kk += 32) {
            BF16Frag a[2];
#pragma unroll
            for (int mi = 0; mi < 2; ++mi) {
                const bf16_t* ar = &sA[(wm0 + mi * 16 + mrow) * kKChunk + kk];
                a[mi].h[0] = *(const v8bf*)(ar + khalf * 16);
                a[mi].h[1] = *(const v8bf*)(ar + khalf * 16 + 8);
            }
            const int k = kc + kk;
#pragma unroll
            for (int ni = 0; ni < 2; ++ni) {
                __builtin_prefetch(bG[ni] + k + 256, 0, 1);
                __builtin_prefetch(bU[ni] + k + 256, 0, 1);
                BF16Frag fg, fu;
                fg.h[0] = *(const v8bf*)(bG[ni] + k);
                fg.h[1] = *(const v8bf*)(bG[ni] + k + 8);
                fu.h[0] = *(const v8bf*)(bU[ni] + k);
                fu.h[1] = *(const v8bf*)(bU[ni] + k + 8);
#pragma unroll
                for (int mi = 0; mi < 2; ++mi) {
                    accG[mi][ni] = wmma_bf16(a[mi], fg, accG[mi][ni]);
                    accU[mi][ni] = wmma_bf16(a[mi], fu, accU[mi][ni]);
                }
            }
        }
        __syncthreads();   // sA reuse fence before next chunk
    }

#pragma unroll
    for (int mi = 0; mi < 2; ++mi) {
#pragma unroll
        for (int ni = 0; ni < 2; ++ni) {
#pragma unroll
            for (int r = 0; r < 8; ++r) {
                const int m = wm0 + mi * 16 + r + khalf * 8;
                if (m < mcnt) {
                    float g = accG[mi][ni][r], u = accU[mi][ni][r];
                    float h = (g / (1.0f + __expf(-g))) * u;   // silu(g)*u
                    hidden[(size_t)(base + m0 + m) * kHid + hg0 + ni * 16 +
                           mrow] = (bf16_t)h;
                }
            }
        }
    }
}

// ---------------------------------------------------------------------------
// Kernel 5b: down GEMM.  expert_out[slot][d] = hidden[slot] @ down[e]^T, f32.
// Same tiling; A rows are contiguous slots (no gather), K = 2816.
// ---------------------------------------------------------------------------
__global__ __launch_bounds__(256) void k_down(
    const bf16_t* __restrict__ hidden, const bf16_t* __restrict__ dn_w,
    const int* __restrict__ counts, const int* __restrict__ offsets,
    float* __restrict__ expert_out) {
    const int e   = blockIdx.z;
    const int mt  = blockIdx.y;
    const int nt  = blockIdx.x;
    const int cnt = counts[e];
    const int m0  = mt * kMTile;
    if (m0 >= cnt) return;
    const int base = offsets[e];
    const int mcnt = min(kMTile, cnt - m0);

    __shared__ bf16_t sA[kMTile * kKChunk];   // 64 KB

    const int tid   = threadIdx.x;
    const int lane  = tid & 31;
    const int wave  = tid >> 5;
    const int waveM = wave >> 1;
    const int waveN = wave & 1;
    const int mrow  = lane & 15;
    const int khalf = lane >> 4;
    const int wm0   = waveM * 32;
    const int d0    = nt * kNTile + waveN * 32;

    const bf16_t* dnE = dn_w + (size_t)e * kDim * kHid;
    const bf16_t* bD[2];
#pragma unroll
    for (int ni = 0; ni < 2; ++ni)
        bD[ni] = dnE + (size_t)(d0 + ni * 16 + mrow) * kHid + khalf * 16;

    const int arow  = tid >> 1;
    const int ahalf = tid & 1;
    const bf16_t* aSrcRow =
        hidden + (size_t)(base + m0 + ((arow < mcnt) ? arow : 0)) * kHid +
        ahalf * (kKChunk / 2);
    const unsigned aLds =
        (unsigned)(uintptr_t)(&sA[arow * kKChunk + ahalf * (kKChunk / 2)]);

    v8f acc[2][2] = {{{}, {}}, {{}, {}}};

    for (int kc = 0; kc < kHid; kc += kKChunk) {   // 11 chunks of 256
        const bf16_t* src = aSrcRow + kc;
#pragma unroll
        for (int i = 0; i < 16; ++i)
            async_copy_b128(aLds + i * 16, src + i * 8);
        async_wait_all();
        __syncthreads();

        for (int kk = 0; kk < kKChunk; kk += 32) {
            BF16Frag a[2];
#pragma unroll
            for (int mi = 0; mi < 2; ++mi) {
                const bf16_t* ar = &sA[(wm0 + mi * 16 + mrow) * kKChunk + kk];
                a[mi].h[0] = *(const v8bf*)(ar + khalf * 16);
                a[mi].h[1] = *(const v8bf*)(ar + khalf * 16 + 8);
            }
            const int k = kc + kk;
#pragma unroll
            for (int ni = 0; ni < 2; ++ni) {
                __builtin_prefetch(bD[ni] + k + 256, 0, 1);
                BF16Frag b;
                b.h[0] = *(const v8bf*)(bD[ni] + k);
                b.h[1] = *(const v8bf*)(bD[ni] + k + 8);
#pragma unroll
                for (int mi = 0; mi < 2; ++mi)
                    acc[mi][ni] = wmma_bf16(a[mi], b, acc[mi][ni]);
            }
        }
        __syncthreads();
    }

#pragma unroll
    for (int mi = 0; mi < 2; ++mi) {
#pragma unroll
        for (int ni = 0; ni < 2; ++ni) {
#pragma unroll
            for (int r = 0; r < 8; ++r) {
                const int m = wm0 + mi * 16 + r + khalf * 8;
                if (m < mcnt)
                    expert_out[(size_t)(base + m0 + m) * kDim + d0 + ni * 16 +
                               mrow] = acc[mi][ni][r];
            }
        }
    }
}

// ---------------------------------------------------------------------------
// Kernel 6: deterministic combine: out = x + clip(alpha) * (w0*eo0 + w1*eo1)
// ---------------------------------------------------------------------------
__global__ void k_combine(const float* __restrict__ x,
                          const float* __restrict__ expert_out,
                          const int* __restrict__ slot_of,
                          const float* __restrict__ topk_w,
                          const float* __restrict__ alpha_p,
                          float* __restrict__ out) {
    size_t i = (size_t)blockIdx.x * blockDim.x + threadIdx.x;
    if (i >= (size_t)kTokens * kDim) return;
    const int t = (int)(i >> 10);
    const int d = (int)(i & (kDim - 1));
    const float alpha = fminf(fmaxf(alpha_p[0], -0.5f), 2.0f);
    const int   s0 = slot_of[t * 2 + 0], s1 = slot_of[t * 2 + 1];
    const float w0 = topk_w[t * 2 + 0],  w1 = topk_w[t * 2 + 1];
    const float r  = w0 * expert_out[(size_t)s0 * kDim + d] +
                     w1 * expert_out[(size_t)s1 * kDim + d];
    out[i] = x[i] + alpha * r;
}

// ---------------------------------------------------------------------------
// Host-side launcher.
// ---------------------------------------------------------------------------
extern "C" void kernel_launch(void* const* d_in, const int* in_sizes, int n_in,
                              void* d_out, int out_size, void* d_ws,
                              size_t ws_size, hipStream_t stream) {
    const float* x        = (const float*)d_in[0];
    const float* norm_w   = (const float*)d_in[1];
    const float* router_w = (const float*)d_in[2];
    const float* gu_f32   = (const float*)d_in[3];
    const float* dn_f32   = (const float*)d_in[4];
    const float* alpha    = (const float*)d_in[5];
    float*       out      = (float*)d_out;

    char*  ws  = (char*)d_ws;
    size_t off = 0;
    auto alloc = [&](size_t bytes) -> char* {
        char* p = ws + off;
        off = (off + bytes + 255) & ~(size_t)255;
        return p;
    };

    bf16_t* normed     = (bf16_t*)alloc((size_t)kTokens * kDim * sizeof(bf16_t));
    bf16_t* gu_bf      = (bf16_t*)alloc((size_t)kE * 2 * kHid * kDim * sizeof(bf16_t));
    bf16_t* dn_bf      = (bf16_t*)alloc((size_t)kE * kDim * kHid * sizeof(bf16_t));
    bf16_t* hidden     = (bf16_t*)alloc((size_t)kSlots * kHid * sizeof(bf16_t));
    int*    counts     = (int*)alloc(kE * sizeof(int));
    int*    offsets    = (int*)alloc(kE * sizeof(int));
    int*    cursors    = (int*)alloc(kE * sizeof(int));
    int*    topk_e     = (int*)alloc((size_t)kTokens * 2 * sizeof(int));
    float*  topk_w     = (float*)alloc((size_t)kTokens * 2 * sizeof(float));
    int*    slot_of    = (int*)alloc((size_t)kTokens * 2 * sizeof(int));
    int*    tok_idx    = (int*)alloc((size_t)kSlots * sizeof(int));
    float*  expert_out = (float*)alloc((size_t)kSlots * kDim * sizeof(float));
    (void)ws_size; (void)in_sizes; (void)n_in; (void)out_size;

    // 0) zero expert counters
    k_zero_counts<<<1, 32, 0, stream>>>(counts);

    // 1) weight conversion f32 -> bf16 (139 MB bf16 -> L2 resident)
    const long n4_gu = (long)kE * 2 * kHid * kDim / 4;
    const long n4_dn = (long)kE * kDim * kHid / 4;
    k_cvt_f32_bf16<<<2048, 256, 0, stream>>>(gu_f32, gu_bf, n4_gu);
    k_cvt_f32_bf16<<<2048, 256, 0, stream>>>(dn_f32, dn_bf, n4_dn);

    // 2) fused RMSNorm + router top-2
    k_rmsnorm_router<<<kTokens, 256, 0, stream>>>(x, norm_w, router_w, normed,
                                                  topk_e, topk_w, counts);

    // 3) prefix scan + 4) scatter into contiguous per-expert slot ranges
    k_scan_offsets<<<1, 32, 0, stream>>>(counts, offsets, cursors);
    k_scatter<<<kTokens / 256, 256, 0, stream>>>(topk_e, cursors, offsets,
                                                 tok_idx, slot_of);

    // 5a) gate_up GEMM + silu  (grid: N tiles x M tiles x experts)
    dim3 gA(kHid / kNTile, kMTiles, kE);    // 44 x 64 x 8
    k_gateup_silu<<<gA, 256, 0, stream>>>(normed, gu_bf, tok_idx, counts,
                                          offsets, hidden);

    // 5b) down GEMM
    dim3 gB(kDim / kNTile, kMTiles, kE);    // 16 x 64 x 8
    k_down<<<gB, 256, 0, stream>>>(hidden, dn_bf, counts, offsets, expert_out);

    // 6) residual + weighted combine (deterministic, atomic-free)
    k_combine<<<(kTokens * kDim) / 256, 256, 0, stream>>>(
        x, expert_out, slot_of, topk_w, alpha, out);
}
